// VAE_LSTM_36584531427456
// MI455X (gfx1250) — compile-verified, matrix operand order checked by reading, not run on hardware
//
#include <hip/hip_runtime.h>
#include <hip/hip_bf16.h>

// ---------------------------------------------------------------------------
// VAE-LSTM forward for MI455X (gfx1250, wave32, WMMA).
// B=2048 I=512 H=512 L=32 D=1024 O=512.
// - straight-through z == one_hot(argmax(pre+g)) in forward => t2h matmuls
//   are row gathers of t2h_W^T.
// - GEMMs: bf16 WMMA (v_wmma_f32_16x16x32_bf16), fp32 accumulate.
// - Global->LDS staging uses CDNA5 async-to-LDS (ASYNCcnt) with a double
//   buffer when the builtins are available; sync fallback otherwise.
// ---------------------------------------------------------------------------

typedef __attribute__((ext_vector_type(16))) __bf16 v16bf;
typedef __attribute__((ext_vector_type(8)))  __bf16 v8bf;
typedef __attribute__((ext_vector_type(8)))  float  v8f;
typedef int v4i __attribute__((vector_size(16)));

#define B_   2048
#define I_   512
#define H_   512
#define L_   32
#define D_   1024
#define O_   512

#if defined(__has_builtin)
#if __has_builtin(__builtin_amdgcn_global_load_async_to_lds_b128)
#define ASYNC_LDS 1
#endif
#endif

#ifdef ASYNC_LDS
typedef __attribute__((address_space(1))) v4i* gv4i_p;  // global v4i*
typedef __attribute__((address_space(3))) v4i* lv4i_p;  // LDS v4i*
#endif

__device__ __forceinline__ void cp16(__bf16* lds, const __bf16* g) {
#ifdef ASYNC_LDS
  __builtin_amdgcn_global_load_async_to_lds_b128((gv4i_p)g, (lv4i_p)lds, 0, 0);
#else
  *(float4*)lds = *(const float4*)g;
#endif
}

template <int N>
__device__ __forceinline__ void async_wait() {
#ifdef ASYNC_LDS
#if __has_builtin(__builtin_amdgcn_s_wait_asynccnt)
  __builtin_amdgcn_s_wait_asynccnt(N);
#else
  asm volatile("s_wait_asynccnt %0" ::"i"(N) : "memory");
#endif
#endif
}

// ---------------- prep kernels ----------------
__global__ void k_cvt_bf16(const float* __restrict__ src, __bf16* __restrict__ dst, int n) {
  int i = blockIdx.x * blockDim.x + threadIdx.x;
  if (i < n) dst[i] = (__bf16)src[i];
}

// dst[r][0:k) = A[r][:], dst[r][k:2k) = Bm[r][:]   (concat Wih|Whh per row)
__global__ void k_concat_bf16(const float* __restrict__ A, const float* __restrict__ Bm,
                              __bf16* __restrict__ dst, int rows, int k) {
  int i = blockIdx.x * blockDim.x + threadIdx.x;
  int total = rows * 2 * k;
  if (i >= total) return;
  int r = i / (2 * k), c = i - r * 2 * k;
  float v = (c < k) ? A[(size_t)r * k + c] : Bm[(size_t)r * k + (c - k)];
  dst[i] = (__bf16)v;
}

// dst[c][r] = src[r][c]   (t2h_W [H,D] -> t2h_WT [D,H], row-gatherable)
__global__ void k_transpose(const float* __restrict__ src, float* __restrict__ dst,
                            int rows, int cols) {
  int i = blockIdx.x * blockDim.x + threadIdx.x;
  if (i >= rows * cols) return;
  int r = i / cols, c = i - r * cols;
  dst[(size_t)c * rows + r] = src[i];
}

// ---------------- bf16 WMMA GEMM: C[M,N] = A[M,K] @ W[N,K]^T + b1 (+ b2) ----
// Block tile 128x128, 256 threads = 8 waves in 2(M) x 4(N); each wave owns a
// 64x32 tile = 4x2 v_wmma_f32_16x16x32_bf16 accumulators. Double-buffered
// LDS staging via GLOBAL_LOAD_ASYNC_TO_LDS_B128 (ASYNCcnt) when available.
__global__ __launch_bounds__(256)
void k_gemm_bf16(const __bf16* __restrict__ A, int lda,
                 const __bf16* __restrict__ W, int ldw,
                 const float* __restrict__ b1, const float* __restrict__ b2,
                 float* __restrict__ C, long ldc, int K) {
  __shared__ __bf16 lA[2][128 * 40];   // 40 = 32 + 8 pad (80B row, 16B aligned)
  __shared__ __bf16 lW[2][128 * 40];

  const int tid  = threadIdx.x;
  const int lane = tid & 31;
  const int wave = tid >> 5;
  const int wm   = wave & 1;        // M offset 0/64
  const int wn   = wave >> 1;       // N offset 0/32/64/96
  const long mBlk = (long)blockIdx.y * 128;
  const long nBlk = (long)blockIdx.x * 128;
  const int lrow = lane & 15;
  const int koff = (lane >> 4) * 8; // ISA 16-bit A layout: lanes 0-15 K0..7/16..23,
                                    // lanes 16-31 K8..15/24..31
  v8f acc[4][2];
#pragma unroll
  for (int m = 0; m < 4; ++m)
#pragma unroll
    for (int n = 0; n < 2; ++n)
#pragma unroll
      for (int i = 0; i < 8; ++i) acc[m][n][i] = 0.f;

  // stage one 128x32 tile of A and W into LDS buffer `buf`
  // (per wave: 4 async-to-LDS instructions => s_wait_asynccnt 4 retires the
  //  previous stage while this one is in flight)
  auto stage = [&](int buf, int k0) {
#pragma unroll
    for (int c = tid; c < 512; c += 256) {
      int row = c >> 2, col = (c & 3) << 3;
      cp16(&lA[buf][row * 40 + col], &A[(mBlk + row) * lda + k0 + col]);
      cp16(&lW[buf][row * 40 + col], &W[(nBlk + row) * ldw + k0 + col]);
    }
  };

  const int nk = K >> 5;
  stage(0, 0);
  for (int kb = 0; kb < nk; ++kb) {
    const int buf = kb & 1;
    if (kb + 1 < nk) {
      stage(buf ^ 1, (kb + 1) << 5);   // prefetch next tile (other buffer)
      async_wait<4>();                 // current tile's 4 async ops done
    } else {
      async_wait<0>();
    }
    __syncthreads();                   // publish LDS across waves

    v16bf aF[4], bF[2];
#pragma unroll
    for (int m = 0; m < 4; ++m) {
      int r = wm * 64 + m * 16 + lrow;
      v8bf lo = *(const v8bf*)(&lA[buf][r * 40 + koff]);       // ds_load_b128
      v8bf hi = *(const v8bf*)(&lA[buf][r * 40 + koff + 16]);  // ds_load_b128
#pragma unroll
      for (int i = 0; i < 8; ++i) { aF[m][i] = lo[i]; aF[m][i + 8] = hi[i]; }
    }
#pragma unroll
    for (int n = 0; n < 2; ++n) {
      int r = wn * 32 + n * 16 + lrow;
      v8bf lo = *(const v8bf*)(&lW[buf][r * 40 + koff]);
      v8bf hi = *(const v8bf*)(&lW[buf][r * 40 + koff + 16]);
#pragma unroll
      for (int i = 0; i < 8; ++i) { bF[n][i] = lo[i]; bF[n][i + 8] = hi[i]; }
    }
#pragma unroll
    for (int m = 0; m < 4; ++m)
#pragma unroll
      for (int n = 0; n < 2; ++n)
        acc[m][n] = __builtin_amdgcn_wmma_f32_16x16x32_bf16(
            false, aF[m], false, bF[n], (short)0, acc[m][n], false, false);

    __syncthreads();                   // all waves done reading buf
  }

  // epilogue: ISA C/D layout — lanes 0-15 rows r, lanes 16-31 rows r+8
  const int rb = (lane >> 4) * 8;
#pragma unroll
  for (int n = 0; n < 2; ++n) {
    long col = nBlk + wn * 32 + n * 16 + (lane & 15);
    float bias = b1 ? b1[col] : 0.f;
    if (b2) bias += b2[col];
#pragma unroll
    for (int m = 0; m < 4; ++m) {
      long rowb = mBlk + wm * 64 + m * 16 + rb;
#pragma unroll
      for (int v = 0; v < 8; ++v)
        C[(rowb + v) * ldc + col] = acc[m][n][v] + bias;
    }
  }
}

// ---------------- LSTM pointwise cell ----------------
__global__ void k_lstm_cell(const float* __restrict__ gates, float* __restrict__ c,
                            __bf16* __restrict__ X) {
  int idx = blockIdx.x * blockDim.x + threadIdx.x;
  if (idx >= B_ * H_) return;
  int b = idx >> 9, j = idx & (H_ - 1);
  const float* g = gates + (size_t)b * 4 * H_;
  float ig = 1.f / (1.f + __expf(-g[j]));
  float fg = 1.f / (1.f + __expf(-g[H_ + j]));
  float gg = tanhf(g[2 * H_ + j]);
  float og = 1.f / (1.f + __expf(-g[3 * H_ + j]));
  float cn = fg * c[idx] + ig * gg;
  c[idx] = cn;
  X[(size_t)b * (2 * H_) + H_ + j] = (__bf16)(og * tanhf(cn));
}

// ---------------- argmax + one-hot + messages + next-input gather ----------
__global__ __launch_bounds__(256)
void k_argmax_st(const float* __restrict__ logits_t, long lg_stride,
                 const float* __restrict__ gum_t,
                 const float* __restrict__ t2h_WT, const float* __restrict__ t2h_b,
                 float* __restrict__ onehot_t, long oh_stride,
                 float* __restrict__ msg, int t,
                 int* __restrict__ amax_buf, __bf16* __restrict__ X) {
  __shared__ float smax[256];
  __shared__ int   sidx[256];
  int b = blockIdx.x, tid = threadIdx.x;
  const float* lg = logits_t + (long)b * lg_stride;
  const float* gn = gum_t + (long)b * D_;
  float best = -3.4e38f; int bi = 0;
  for (int d = tid; d < D_; d += 256) {
    float v = lg[d] + gn[d];
    if (v > best) { best = v; bi = d; }
  }
  smax[tid] = best; sidx[tid] = bi;
  __syncthreads();
  for (int s = 128; s > 0; s >>= 1) {
    if (tid < s) {
      float o = smax[tid + s]; int oi = sidx[tid + s];
      if (o > smax[tid] || (o == smax[tid] && oi < sidx[tid])) {
        smax[tid] = o; sidx[tid] = oi;
      }
    }
    __syncthreads();
  }
  int idx = sidx[0];
  float* oh = onehot_t + (long)b * oh_stride;
  for (int d = tid; d < D_; d += 256) oh[d] = (d == idx) ? 1.f : 0.f;
  if (tid == 0) { msg[(long)b * L_ + t] = (float)idx; amax_buf[b] = idx; }
  const float* wr = t2h_WT + (long)idx * H_;
  for (int j = tid; j < H_; j += 256)
    X[(long)b * (2 * H_) + j] = (__bf16)(wr[j] + t2h_b[j]);
}

// decoder input embedding: X[b][0:H) = t2h_WT[amax[b]] + t2h_b
__global__ void k_gather_emb(const int* __restrict__ amax, const float* __restrict__ t2h_WT,
                             const float* __restrict__ t2h_b, __bf16* __restrict__ X) {
  int i = blockIdx.x * blockDim.x + threadIdx.x;
  if (i >= B_ * H_) return;
  int b = i >> 9, j = i & (H_ - 1);
  int idx = amax[b];
  X[(long)b * (2 * H_) + j] = (__bf16)(t2h_WT[(long)idx * H_ + j] + t2h_b[j]);
}

// ---------------- host driver ----------------
static inline void launch_gemm(const __bf16* A, int lda, const __bf16* W, int ldw,
                               const float* b1, const float* b2,
                               float* C, long ldc, int M, int N, int K,
                               hipStream_t s) {
  dim3 g(N / 128, M / 128);
  k_gemm_bf16<<<g, 256, 0, s>>>(A, lda, W, ldw, b1, b2, C, ldc, K);
}

extern "C" void kernel_launch(void* const* d_in, const int* in_sizes, int n_in,
                              void* d_out, int out_size, void* d_ws, size_t ws_size,
                              hipStream_t stream) {
  (void)in_sizes; (void)n_in; (void)out_size; (void)ws_size;
  const float* x       = (const float*)d_in[0];
  const float* gumbel  = (const float*)d_in[1];
  const float* in_W    = (const float*)d_in[2];
  const float* in_b    = (const float*)d_in[3];
  const float* out_W   = (const float*)d_in[4];
  const float* out_b   = (const float*)d_in[5];
  const float* enc_Wih = (const float*)d_in[6];
  const float* enc_Whh = (const float*)d_in[7];
  const float* enc_bih = (const float*)d_in[8];
  const float* enc_bhh = (const float*)d_in[9];
  const float* dec_Wih = (const float*)d_in[10];
  const float* dec_Whh = (const float*)d_in[11];
  const float* dec_bih = (const float*)d_in[12];
  const float* dec_bhh = (const float*)d_in[13];
  const float* h2t_W   = (const float*)d_in[14];
  const float* h2t_b   = (const float*)d_in[15];
  const float* t2h_W   = (const float*)d_in[16];
  const float* t2h_b   = (const float*)d_in[17];

  float* recon  = (float*)d_out;                      // [B,O]
  float* onehot = recon  + (size_t)B_ * O_;           // [B,L,D]
  float* logits = onehot + (size_t)B_ * L_ * D_;      // [B,L,D]
  float* msg    = logits + (size_t)B_ * L_ * D_;      // [B,L]

  // workspace layout (~39 MB)
  char* ws = (char*)d_ws;
  size_t off = 0;
  auto alloc = [&](size_t bytes) -> void* {
    void* p = ws + off; off += (bytes + 255) & ~(size_t)255; return p;
  };
  __bf16* W_enc   = (__bf16*)alloc((size_t)4 * H_ * 2 * H_ * 2); // [4H,2H]
  __bf16* W_dec   = (__bf16*)alloc((size_t)4 * H_ * 2 * H_ * 2);
  __bf16* W_in    = (__bf16*)alloc((size_t)H_ * I_ * 2);
  __bf16* W_out   = (__bf16*)alloc((size_t)O_ * H_ * 2);
  __bf16* W_h2t   = (__bf16*)alloc((size_t)D_ * H_ * 2);
  float*  t2h_WT  = (float*) alloc((size_t)D_ * H_ * 4);
  __bf16* x_bf    = (__bf16*)alloc((size_t)B_ * I_ * 2);
  __bf16* X       = (__bf16*)alloc((size_t)B_ * 2 * H_ * 2);     // [B, inp|h]
  float*  gatesb  = (float*) alloc((size_t)B_ * 4 * H_ * 4);
  float*  cbuf    = (float*) alloc((size_t)B_ * H_ * 4);
  int*    amaxbuf = (int*)   alloc((size_t)L_ * B_ * 4);

  const int T256 = 256;
  // ---- weight prep (bf16 conversion + concat + transpose) ----
  k_concat_bf16<<<(4 * H_ * 2 * H_ + T256 - 1) / T256, T256, 0, stream>>>(
      enc_Wih, enc_Whh, W_enc, 4 * H_, H_);
  k_concat_bf16<<<(4 * H_ * 2 * H_ + T256 - 1) / T256, T256, 0, stream>>>(
      dec_Wih, dec_Whh, W_dec, 4 * H_, H_);
  k_cvt_bf16<<<(H_ * I_ + T256 - 1) / T256, T256, 0, stream>>>(in_W, W_in, H_ * I_);
  k_cvt_bf16<<<(O_ * H_ + T256 - 1) / T256, T256, 0, stream>>>(out_W, W_out, O_ * H_);
  k_cvt_bf16<<<(D_ * H_ + T256 - 1) / T256, T256, 0, stream>>>(h2t_W, W_h2t, D_ * H_);
  k_cvt_bf16<<<(B_ * I_ + T256 - 1) / T256, T256, 0, stream>>>(x, x_bf, B_ * I_);
  k_transpose<<<(H_ * D_ + T256 - 1) / T256, T256, 0, stream>>>(t2h_W, t2h_WT, H_, D_);

  // ---- encoder init: c0 = x_emb = x @ in_W^T + in_b ; h0 = 0 ; inp0 = 0 ----
  launch_gemm(x_bf, I_, W_in, I_, in_b, nullptr, cbuf, H_, B_, H_, I_, stream);
  (void)hipMemsetAsync(X, 0, (size_t)B_ * 2 * H_ * 2, stream);

  // ---- encoder scan ----
  for (int t = 0; t < L_; ++t) {
    launch_gemm(X, 2 * H_, W_enc, 2 * H_, enc_bih, enc_bhh,
                gatesb, 4 * H_, B_, 4 * H_, 2 * H_, stream);
    k_lstm_cell<<<(B_ * H_ + T256 - 1) / T256, T256, 0, stream>>>(gatesb, cbuf, X);
    launch_gemm(X + H_, 2 * H_, W_h2t, H_, h2t_b, nullptr,
                logits + (size_t)t * D_, (long)L_ * D_, B_, D_, H_, stream);
    k_argmax_st<<<B_, 256, 0, stream>>>(
        logits + (size_t)t * D_, (long)L_ * D_,
        gumbel + (size_t)t * B_ * D_,
        t2h_WT, t2h_b,
        onehot + (size_t)t * D_, (long)L_ * D_,
        msg, t, amaxbuf + (size_t)t * B_, X);
  }

  // ---- decoder init: h=0, c=0 ----
  (void)hipMemsetAsync(X, 0, (size_t)B_ * 2 * H_ * 2, stream);
  (void)hipMemsetAsync(cbuf, 0, (size_t)B_ * H_ * 4, stream);

  // ---- decoder scan ----
  for (int t = 0; t < L_; ++t) {
    k_gather_emb<<<(B_ * H_ + T256 - 1) / T256, T256, 0, stream>>>(
        amaxbuf + (size_t)t * B_, t2h_WT, t2h_b, X);
    launch_gemm(X, 2 * H_, W_dec, 2 * H_, dec_bih, dec_bhh,
                gatesb, 4 * H_, B_, 4 * H_, 2 * H_, stream);
    k_lstm_cell<<<(B_ * H_ + T256 - 1) / T256, T256, 0, stream>>>(gatesb, cbuf, X);
  }

  // ---- recon = h_final @ out_W^T + out_b ----
  launch_gemm(X + H_, 2 * H_, W_out, H_, out_b, nullptr,
              recon, O_, B_, O_, H_, stream);
}
